// MambaForecaster_89842125898128
// MI455X (gfx1250) — matrix operand order repaired and checked by reading, not compiled, not run on hardware
//
#include <hip/hip_runtime.h>
#include <hip/hip_bf16.h>
#include <math.h>

// ---------------------------------------------------------------------------
// MambaForecaster forward for MI455X (gfx1250, wave32).
// Dense einsums -> V_WMMA_F32_16X16X32_F16, one 32x32 output tile per wave.
// All GEMM operands are pre-converted to f16 (weights once per launch,
// activations at their producers), so the GEMM inner loop is just
// 8x global_load_b128 + 4x v_wmma per k-step: no cvts, no guards.
// ---------------------------------------------------------------------------

#define D_MODEL 128
#define N_LAYERS 4
#define D_STATE 16
#define D_CONV 4
#define D_INNER 256
#define DT_RANK 8
#define SEQ 512
#define PATCH 16
#define N_PATCH 32
#define PRED 96
#define COMP 64
#define BATCH 512
#define BL (BATCH * N_PATCH)   // 16384 token rows

typedef __attribute__((ext_vector_type(16))) _Float16 v16h;
typedef __attribute__((ext_vector_type(8)))  _Float16 half8;
typedef __attribute__((ext_vector_type(8)))  float    v8f;

// ---------------- WMMA fragment loaders (ISA 7.12.2 layouts) ----------------
// A 16x32 f16: lane l -> m=l%16, hi=l/16. elems 0..7 <- k=hi*8+0..7,
// elems 8..15 <- k=16+hi*8+0..7. Two 16B loads.
__device__ inline v16h load_a_frag(const _Float16* __restrict__ arow, int hi) {
    half8 q0 = *(const half8*)(arow + hi * 8);
    half8 q1 = *(const half8*)(arow + 16 + hi * 8);
    v16h a;
#pragma unroll
    for (int j = 0; j < 8; ++j) { a[j] = q0[j]; a[8 + j] = q1[j]; }
    return a;
}

// B 32x16 f16 (KxN) from weight row W[n,:] (N x K row-major):
// lane l -> n=l%16, hi=l/16 ; elems 0..7 <- k=hi*8+0..7, elems 8..15 <- k=16+hi*8+0..7. Two 16B loads.
__device__ inline v16h load_b_frag(const _Float16* __restrict__ wrow, int hi) {
    half8 q0 = *(const half8*)(wrow + hi * 8);
    half8 q1 = *(const half8*)(wrow + 16 + hi * 8);
    v16h b;
#pragma unroll
    for (int j = 0; j < 8; ++j) { b[j] = q0[j]; b[8 + j] = q1[j]; }
    return b;
}

// ---------------- GEMM: C[m,n] = act( sum_k A[m,k]*W[n,k] + bias[n] (+C) )
// One wave per 32x32 tile (2x2 WMMA). Requires M%32==0, N%32==0, K%32==0.
// outmode: 0 = f32 store, 1 = f32 accumulate (residual), 2 = f16 store.
__global__ void gemm_wmma_kernel(const _Float16* __restrict__ A,
                                 const _Float16* __restrict__ W,
                                 const float* __restrict__ bias,
                                 float* __restrict__ C,
                                 _Float16* __restrict__ C16,
                                 int M, int N, int K, int ldc, int tilesN,
                                 int outmode, int act) {
    int wave = blockIdx.x * (blockDim.x >> 5) + (threadIdx.x >> 5);
    int tiles_total = (M >> 5) * tilesN;
    if (wave >= tiles_total) return;
    int tm = wave / tilesN;
    int tn = wave % tilesN;

    int lane = threadIdx.x & 31;
    int l16  = lane & 15;
    int hi   = lane >> 4;

    const _Float16* arow0 = A + (size_t)(tm * 32 + l16) * K;
    const _Float16* arow1 = arow0 + (size_t)16 * K;
    const _Float16* wrow0 = W + (size_t)(tn * 32 + l16) * K;
    const _Float16* wrow1 = wrow0 + (size_t)16 * K;

    v8f acc00 = {}, acc01 = {}, acc10 = {}, acc11 = {};
    for (int kk = 0; kk < K; kk += 32) {
        v16h a0 = load_a_frag(arow0 + kk, hi);
        v16h a1 = load_a_frag(arow1 + kk, hi);
        v16h b0 = load_b_frag(wrow0 + kk, hi);
        v16h b1 = load_b_frag(wrow1 + kk, hi);
        acc00 = __builtin_amdgcn_wmma_f32_16x16x32_f16(false, a0, false, b0, (short)0, acc00, false, false);
        acc01 = __builtin_amdgcn_wmma_f32_16x16x32_f16(false, a0, false, b1, (short)0, acc01, false, false);
        acc10 = __builtin_amdgcn_wmma_f32_16x16x32_f16(false, a1, false, b0, (short)0, acc10, false, false);
        acc11 = __builtin_amdgcn_wmma_f32_16x16x32_f16(false, a1, false, b1, (short)0, acc11, false, false);
    }

    // D layout: VGPR r -> M = r + 8*(lane/16), N = lane%16 (ISA 7.12.2).
    v8f accs[2][2] = { { acc00, acc01 }, { acc10, acc11 } };
#pragma unroll
    for (int mi = 0; mi < 2; ++mi) {
#pragma unroll
        for (int ni = 0; ni < 2; ++ni) {
            int n = tn * 32 + ni * 16 + l16;
            float bv = bias ? bias[n] : 0.0f;
#pragma unroll
            for (int r = 0; r < 8; ++r) {
                int m = tm * 32 + mi * 16 + hi * 8 + r;
                size_t idx = (size_t)m * ldc + n;
                float v = accs[mi][ni][r] + bv;
                if (outmode == 1) v += C[idx];
                if (act == 1) v = 0.5f * v * (1.0f + erff(v * 0.70710678118654752f));
                if (outmode == 2) C16[idx] = (_Float16)v;
                else              C[idx]   = v;
            }
        }
    }
}

// ---------------- weight conversion / padding ----------------
__global__ void cvt_f16_kernel(const float* __restrict__ src,
                               _Float16* __restrict__ dst, int n) {
    int i = blockIdx.x * blockDim.x + threadIdx.x;
    if (i < n) dst[i] = (_Float16)src[i];
}

__global__ void pad2d_f16_kernel(const float* __restrict__ src,
                                 _Float16* __restrict__ dst,
                                 int Rs, int Cs, int Rd, int Cd) {
    int i = blockIdx.x * blockDim.x + threadIdx.x;
    if (i >= Rd * Cd) return;
    int r = i / Cd, c = i % Cd;
    dst[i] = (r < Rs && c < Cs) ? (_Float16)src[r * Cs + c] : (_Float16)0.0f;
}

// ---------------- elementwise kernels ----------------

// xc = x[:,:,-1] / sqrt(1+1e-5) * bn_g + bn_b -> f16 A matrix (BL, 32),
// cols 16..31 zero (patch GEMM runs with K=32).
__global__ void prep_xc_kernel(const float* __restrict__ x,
                               const float* __restrict__ bn_g,
                               const float* __restrict__ bn_b,
                               _Float16* __restrict__ apad, int n) {
    int i = blockIdx.x * blockDim.x + threadIdx.x;
    if (i >= n) return;
    float v = x[(size_t)i * 7 + 6] * (bn_g[0] * 0.9999950000374997f) + bn_b[0];
    int row = i >> 4, col = i & 15;
    apad[(size_t)row * 32 + col] = (_Float16)v;
    apad[(size_t)row * 32 + 16 + col] = (_Float16)0.0f;
}

// reverse patch order per batch, f32 -> f32
__global__ void reverse_patch_kernel(const float* __restrict__ in,
                                     float* __restrict__ out) {
    int i = blockIdx.x * blockDim.x + threadIdx.x;
    if (i >= BL * D_MODEL) return;
    int c = i % D_MODEL;
    int p = (i / D_MODEL) % N_PATCH;
    int b = i / (D_MODEL * N_PATCH);
    out[((size_t)(b * N_PATCH + (N_PATCH - 1 - p))) * D_MODEL + c] = in[i];
}

// reverse patch order per batch, f32 -> f16 (bo for the ta GEMM)
__global__ void reverse_patch_f16_kernel(const float* __restrict__ in,
                                         _Float16* __restrict__ out) {
    int i = blockIdx.x * blockDim.x + threadIdx.x;
    if (i >= BL * D_MODEL) return;
    int c = i % D_MODEL;
    int p = (i / D_MODEL) % N_PATCH;
    int b = i / (D_MODEL * N_PATCH);
    out[((size_t)(b * N_PATCH + (N_PATCH - 1 - p))) * D_MODEL + c] = (_Float16)in[i];
}

// per-row layernorm over D_MODEL=128, f32 in -> f16 out
__global__ void layernorm_kernel(const float* __restrict__ x,
                                 const float* __restrict__ w,
                                 const float* __restrict__ b,
                                 _Float16* __restrict__ out, int rows) {
    int r = blockIdx.x * blockDim.x + threadIdx.x;
    if (r >= rows) return;
    const float* xr = x + (size_t)r * D_MODEL;
    float mu = 0.0f;
#pragma unroll 4
    for (int i = 0; i < D_MODEL; ++i) mu += xr[i];
    mu *= (1.0f / D_MODEL);
    float var = 0.0f;
#pragma unroll 4
    for (int i = 0; i < D_MODEL; ++i) { float d = xr[i] - mu; var += d * d; }
    var *= (1.0f / D_MODEL);
    float inv = rsqrtf(var + 1e-5f);
    _Float16* orow = out + (size_t)r * D_MODEL;
#pragma unroll 4
    for (int i = 0; i < D_MODEL; ++i)
        orow[i] = (_Float16)((xr[i] - mu) * inv * w[i] + b[i]);
}

// depthwise causal conv (width 4, left pad 3) + bias + SiLU.
// xz is f16 (BL, 2*D_INNER); thread per (batch, channel); f16 out.
__global__ void conv_silu_kernel(const _Float16* __restrict__ xz,
                                 const float* __restrict__ conv_w,
                                 const float* __restrict__ conv_b,
                                 _Float16* __restrict__ xc) {
    int t = blockIdx.x * blockDim.x + threadIdx.x;
    if (t >= BATCH * D_INNER) return;
    int e = t % D_INNER;
    int b = t / D_INNER;
    float w0 = conv_w[e * D_CONV + 0], w1 = conv_w[e * D_CONV + 1];
    float w2 = conv_w[e * D_CONV + 2], w3 = conv_w[e * D_CONV + 3];
    float bias = conv_b[e];
    float h0 = 0.0f, h1 = 0.0f, h2 = 0.0f;
    for (int l = 0; l < N_PATCH; ++l) {
        float cur = (float)xz[((size_t)(b * N_PATCH + l)) * (2 * D_INNER) + e];
        float s = bias + w0 * h0 + w1 * h1 + w2 * h2 + w3 * cur;
        float y = s / (1.0f + __expf(-s));   // SiLU
        xc[((size_t)(b * N_PATCH + l)) * D_INNER + e] = (_Float16)y;
        h0 = h1; h1 = h2; h2 = cur;
    }
}

// selective scan: thread per (batch, inner-channel). dbl (f32) rows padded
// to 64. xc/z inputs f16, y output f16.
__global__ void scan_kernel(const float* __restrict__ dbl,
                            const float* __restrict__ dt_w,
                            const float* __restrict__ dt_b,
                            const float* __restrict__ Alog,
                            const float* __restrict__ Dp,
                            const _Float16* __restrict__ xc,
                            const _Float16* __restrict__ xz,  // z at column 256+
                            _Float16* __restrict__ ybuf) {
    int t = blockIdx.x * blockDim.x + threadIdx.x;
    if (t >= BATCH * D_INNER) return;
    int e = t % D_INNER;
    int b = t / D_INNER;

    float negA[D_STATE];
#pragma unroll
    for (int n = 0; n < D_STATE; ++n)
        negA[n] = -__expf(Alog[e * D_STATE + n]);
    float dtw[DT_RANK];
#pragma unroll
    for (int r = 0; r < DT_RANK; ++r) dtw[r] = dt_w[e * DT_RANK + r];
    float dtb = dt_b[e];
    float Dv = Dp[e];

    float h[D_STATE];
#pragma unroll
    for (int n = 0; n < D_STATE; ++n) h[n] = 0.0f;

    for (int l = 0; l < N_PATCH; ++l) {
        size_t row = (size_t)(b * N_PATCH + l);
        const float* dr = dbl + row * 64;
        float dtr = dtb;
#pragma unroll
        for (int r = 0; r < DT_RANK; ++r) dtr += dtw[r] * dr[r];
        float dt = (dtr > 20.0f) ? dtr : log1pf(__expf(dtr));   // softplus

        float xcv = (float)xc[row * D_INNER + e];
        float y = 0.0f;
#pragma unroll
        for (int n = 0; n < D_STATE; ++n) {
            float dA = __expf(dt * negA[n]);
            h[n] = dA * h[n] + dt * dr[DT_RANK + n] * xcv;           // B
            y += h[n] * dr[DT_RANK + D_STATE + n];                   // C
        }
        y += Dv * xcv;
        float zv = (float)xz[row * (2 * D_INNER) + D_INNER + e];
        y *= zv / (1.0f + __expf(-zv));                              // * silu(z)
        ybuf[row * D_INNER + e] = (_Float16)y;
    }
}

// ---------------------------------------------------------------------------
extern "C" void kernel_launch(void* const* d_in, const int* in_sizes, int n_in,
                              void* d_out, int out_size, void* d_ws, size_t ws_size,
                              hipStream_t stream) {
    (void)in_sizes; (void)n_in; (void)out_size; (void)ws_size;

    const float* x      = (const float*)d_in[0];
    const float* bn_g   = (const float*)d_in[1];
    const float* bn_b   = (const float*)d_in[2];
    const float* pe_w   = (const float*)d_in[3];
    const float* pe_b   = (const float*)d_in[4];
    const float* ta_w   = (const float*)d_in[27];
    const float* ta_b   = (const float*)d_in[28];
    const float* comp_w = (const float*)d_in[29];
    const float* comp_b = (const float*)d_in[30];
    const float* head_w = (const float*)d_in[31];
    const float* head_b = (const float*)d_in[32];
    float* out = (float*)d_out;

    // --- workspace carve (16B-aligned chunks, all counts multiples of 8) ---
    char* wsb = (char*)d_ws;
    auto alloc_f = [&](size_t n) { float* p = (float*)wsb;    wsb += n * sizeof(float);    return p; };
    auto alloc_h = [&](size_t n) { _Float16* p = (_Float16*)wsb; wsb += n * sizeof(_Float16); return p; };

    float* x_fwd = alloc_f((size_t)BL * D_MODEL);   // forward residual / fo (f32)
    float* x_bwd = alloc_f((size_t)BL * D_MODEL);   // backward residual (f32)
    float* dblb  = alloc_f((size_t)BL * 64);        // xproj output (N pad 64, f32)

    _Float16* A_patch = alloc_h((size_t)BL * 32);           // K-padded patches
    _Float16* pe_h    = alloc_h((size_t)D_MODEL * 32);      // pe_w pad K 16->32
    _Float16* inprj_h = alloc_h((size_t)2 * N_LAYERS * 2 * D_INNER * D_MODEL);
    _Float16* xprj_h  = alloc_h((size_t)2 * N_LAYERS * 64 * D_INNER);  // N pad 40->64
    _Float16* oprj_h  = alloc_h((size_t)2 * N_LAYERS * D_MODEL * D_INNER);
    _Float16* ta_h    = alloc_h((size_t)D_MODEL * D_MODEL);
    _Float16* comp_h  = alloc_h((size_t)COMP * D_MODEL);
    _Float16* head_h  = alloc_h((size_t)PRED * N_PATCH * COMP);
    _Float16* xn_h    = alloc_h((size_t)BL * D_MODEL);      // layernorm out
    _Float16* xz_h    = alloc_h((size_t)BL * 2 * D_INNER);  // inproj out
    _Float16* xc_h    = alloc_h((size_t)BL * D_INNER);      // conv+silu out
    _Float16* ybuf_h  = alloc_h((size_t)BL * D_INNER);      // scan out
    _Float16* bo_h    = alloc_h((size_t)BL * D_MODEL);      // re-reversed bwd out
    _Float16* c_h     = alloc_h((size_t)BL * D_MODEL);      // fo+ta(bo) as f16
    _Float16* cbuf_h  = alloc_h((size_t)BL * COMP);         // gelu(comp) out

    auto gemm = [&](const _Float16* A, const _Float16* W, const float* bias,
                    float* C, _Float16* C16,
                    int M, int N, int K, int ldc, int outmode, int act) {
        int tilesN = N / 32;
        int tiles  = (M / 32) * tilesN;
        int blocks = (tiles + 3) / 4;           // 4 waves (128 threads) per block
        gemm_wmma_kernel<<<blocks, 128, 0, stream>>>(A, W, bias, C, C16,
                                                     M, N, K, ldc, tilesN, outmode, act);
    };
    auto cvt = [&](const float* src, _Float16* dst, int n) {
        cvt_f16_kernel<<<(n + 255) / 256, 256, 0, stream>>>(src, dst, n);
    };

    // --- 0) convert/pad every GEMM weight to f16 once per launch ---
    pad2d_f16_kernel<<<(D_MODEL * 32 + 255) / 256, 256, 0, stream>>>(
        pe_w, pe_h, D_MODEL, PATCH, D_MODEL, 32);
    cvt(ta_w,   ta_h,   D_MODEL * D_MODEL);
    cvt(comp_w, comp_h, COMP * D_MODEL);
    cvt(head_w, head_h, PRED * N_PATCH * COMP);
    for (int dir = 0; dir < 2; ++dir) {
        int base = 5 + 11 * dir;
        const float* inprj = (const float*)d_in[base + 2];
        const float* xprj  = (const float*)d_in[base + 5];
        const float* oprj  = (const float*)d_in[base + 10];
        cvt(inprj, inprj_h + (size_t)dir * N_LAYERS * 2 * D_INNER * D_MODEL,
            N_LAYERS * 2 * D_INNER * D_MODEL);
        cvt(oprj,  oprj_h  + (size_t)dir * N_LAYERS * D_MODEL * D_INNER,
            N_LAYERS * D_MODEL * D_INNER);
        for (int L = 0; L < N_LAYERS; ++L)
            pad2d_f16_kernel<<<(64 * D_INNER + 255) / 256, 256, 0, stream>>>(
                xprj + (size_t)L * (DT_RANK + 2 * D_STATE) * D_INNER,
                xprj_h + (size_t)(dir * N_LAYERS + L) * 64 * D_INNER,
                DT_RANK + 2 * D_STATE, D_INNER, 64, D_INNER);
    }

    // --- 1) channel extract + RevIN-style scale into K-padded f16 A ---
    prep_xc_kernel<<<(BATCH * SEQ + 255) / 256, 256, 0, stream>>>(
        x, bn_g, bn_b, A_patch, BATCH * SEQ);
    // --- 2) patch embedding: (BL,32) x (128,32)^T -> x_fwd (BL,128) f32 ---
    gemm(A_patch, pe_h, pe_b, x_fwd, nullptr, BL, D_MODEL, 32, D_MODEL, 0, 0);
    // --- 3) backward stream = patch-reversed copy ---
    reverse_patch_kernel<<<(BL * D_MODEL + 255) / 256, 256, 0, stream>>>(x_fwd, x_bwd);

    // --- 4) two Mamba encoders ---
    for (int dir = 0; dir < 2; ++dir) {
        int base = 5 + 11 * dir;
        const float* ln_w = (const float*)d_in[base + 0];
        const float* ln_b = (const float*)d_in[base + 1];
        const float* cv_w = (const float*)d_in[base + 3];
        const float* cv_b = (const float*)d_in[base + 4];
        const float* dt_w = (const float*)d_in[base + 6];
        const float* dt_b = (const float*)d_in[base + 7];
        const float* Alog = (const float*)d_in[base + 8];
        const float* Dp   = (const float*)d_in[base + 9];
        float* xres = dir ? x_bwd : x_fwd;

        for (int L = 0; L < N_LAYERS; ++L) {
            int slot = dir * N_LAYERS + L;
            layernorm_kernel<<<(BL + 255) / 256, 256, 0, stream>>>(
                xres, ln_w + L * D_MODEL, ln_b + L * D_MODEL, xn_h, BL);
            // inproj: (BL,128) x (512,128)^T -> xz f16 (BL,512)
            gemm(xn_h, inprj_h + (size_t)slot * 2 * D_INNER * D_MODEL, nullptr,
                 nullptr, xz_h, BL, 2 * D_INNER, D_MODEL, 2 * D_INNER, 2, 0);
            conv_silu_kernel<<<(BATCH * D_INNER + 255) / 256, 256, 0, stream>>>(
                xz_h, cv_w + (size_t)L * D_INNER * D_CONV, cv_b + L * D_INNER, xc_h);
            // xproj: (BL,256) x (64,256)^T -> dbl f32 (BL,64)
            gemm(xc_h, xprj_h + (size_t)slot * 64 * D_INNER, nullptr,
                 dblb, nullptr, BL, 64, D_INNER, 64, 0, 0);
            scan_kernel<<<(BATCH * D_INNER + 255) / 256, 256, 0, stream>>>(
                dblb, dt_w + (size_t)L * D_INNER * DT_RANK, dt_b + L * D_INNER,
                Alog + (size_t)L * D_INNER * D_STATE, Dp + L * D_INNER,
                xc_h, xz_h, ybuf_h);
            // outproj accumulated into residual: xres += y @ outproj^T
            gemm(ybuf_h, oprj_h + (size_t)slot * D_MODEL * D_INNER, nullptr,
                 xres, nullptr, BL, D_MODEL, D_INNER, D_MODEL, 1, 0);
        }
    }

    // --- 5) un-reverse bwd output, temporal-align, combine, compress, head ---
    reverse_patch_f16_kernel<<<(BL * D_MODEL + 255) / 256, 256, 0, stream>>>(x_bwd, bo_h);
    // x_fwd (fo) += bo @ ta_w^T + ta_b
    gemm(bo_h, ta_h, ta_b, x_fwd, nullptr, BL, D_MODEL, D_MODEL, D_MODEL, 1, 0);
    // f16 copy of combined stream for the comp GEMM
    cvt(x_fwd, c_h, BL * D_MODEL);
    // cbuf = gelu(c @ comp_w^T + comp_b) as f16
    gemm(c_h, comp_h, comp_b, nullptr, cbuf_h, BL, COMP, D_MODEL, COMP, 2, 1);
    // out (512,96) = cbuf.reshape(512,2048) @ head_w^T + head_b (f32)
    gemm(cbuf_h, head_h, head_b, out, nullptr, BATCH, PRED, N_PATCH * COMP, PRED, 0, 0);
}
